// GraphFP_48017734369780
// MI455X (gfx1250) — compile-verified, matrix operand order checked by reading, not run on hardware
//
#include <hip/hip_runtime.h>
#include <hip/hip_bf16.h>

#define Nn      50000
#define NODEF   192
#define EDGEF   64
#define Mm      800000
#define DIM     256
#define OUTF    2048
#define NDEPTH  3
#define LLD     (OUTF + 8)   // padded leading dim for logits in LDS

typedef __attribute__((ext_vector_type(16))) __bf16 v16bf;
typedef __attribute__((ext_vector_type(8)))  float  v8f;

#define WMMA_BF16(A, B, C) \
  __builtin_amdgcn_wmma_f32_16x16x32_bf16(false, (A), false, (B), (short)0, (C), false, false)

// ---------------------------------------------------------------------------
// WMMA lane layout (16-bit A 16x32 / B 32x16, wave32):
//   lane 0-15  hold koff {0..7, 16..23}, lane 16-31 hold koff {8..15, 24..31}
//   within a lane's 16 elements: i = (koff&7) + ((koff>>4)<<3)
// A: lanes index rows (M); B: lanes index cols (N).
// ---------------------------------------------------------------------------

// Stage a 16xDIM f32 tile into LDS as bf16 in WMMA A-operand order.
// 256 threads: thread = k column, loops 16 rows (coalesced row reads).
__device__ __forceinline__ void stage_A(const float* __restrict__ src, int row0,
                                        __bf16* ldsA, int tid) {
  int k = tid;                 // 0..255
  int kk = k >> 5;
  int koff = k & 31;
  int i = (koff & 7) + ((koff >> 4) << 3);
  int lane_hi = (koff >> 3) & 1;
  #pragma unroll 4
  for (int r = 0; r < 16; ++r) {
    float val = src[(long)(row0 + r) * DIM + k];
    ldsA[((kk * 32 + lane_hi * 16 + r) << 4) + i] = (__bf16)val;
  }
}

// ---------------------------------------------------------------------------
// Pack f32 weights [n_mats][K][Ncols] into bf16 WMMA-B lane order:
// dst[(((mat*Ncols/16 + ct)*K/32 + kk)*32 + lane)*16 + i]
// ---------------------------------------------------------------------------
__global__ void pack_weight_bf16(const float* __restrict__ W, __bf16* __restrict__ dst,
                                 int K, int Ncols, int n_mats) {
  long idx = (long)blockIdx.x * 256 + threadIdx.x;
  long total = (long)n_mats * (Ncols / 16) * (K / 32) * 512;
  if (idx >= total) return;
  int i    = (int)(idx & 15);
  int lane = (int)((idx >> 4) & 31);
  long rest = idx >> 9;
  int nk = K / 32;
  int kk = (int)(rest % nk);  rest /= nk;
  int nct = Ncols / 16;
  int ct = (int)(rest % nct);
  int d  = (int)(rest / nct);
  int col  = ct * 16 + (lane & 15);
  int koff = (i & 7) + ((lane >> 4) << 3) + ((i >> 3) << 4);
  int k    = kk * 32 + koff;
  dst[idx] = (__bf16)W[((long)d * K + k) * Ncols + col];
}

// ---------------------------------------------------------------------------
// attr[N,256] = [node_attr | zeros(64)]
// ---------------------------------------------------------------------------
__global__ void pad_attr_kernel(const float* __restrict__ node_attr,
                                float* __restrict__ attr) {
  long idx = (long)blockIdx.x * 256 + threadIdx.x;   // float4 index
  if (idx >= (long)Nn * (DIM / 4)) return;
  long node = idx >> 6;
  int c4 = (int)(idx & 63);
  float4 v = make_float4(0.f, 0.f, 0.f, 0.f);
  if (c4 < NODEF / 4)
    v = ((const float4*)node_attr)[node * (NODEF / 4) + c4];
  ((float4*)attr)[idx] = v;
}

__global__ void copy_f32x4_kernel(const float* __restrict__ src, float* __restrict__ dst,
                                  long n4) {
  long idx = (long)blockIdx.x * 256 + threadIdx.x;
  if (idx < n4) ((float4*)dst)[idx] = ((const float4*)src)[idx];
}

__global__ void zero_kernel(float* __restrict__ p, int n) {
  int idx = blockIdx.x * 256 + threadIdx.x;
  if (idx < n) p[idx] = 0.f;
}

// ---------------------------------------------------------------------------
// Message passing: v[dst] += attr[src] + [0 | edge_attr].  64 lanes per edge,
// float4 per lane.  attr (51MB) is L2-resident on MI455X (192MB L2).
// ---------------------------------------------------------------------------
__global__ void scatter_edges_kernel(const float* __restrict__ attr,
                                     const float* __restrict__ edge_attr,
                                     const int* __restrict__ esrc,
                                     const int* __restrict__ edst,
                                     float* __restrict__ v) {
  long t = (long)blockIdx.x * 256 + threadIdx.x;
  long e = t >> 6;
  if (e >= Mm) return;
  int j4 = (int)(t & 63) * 4;
  int s = esrc[e], d = edst[e];
  float4 val = *(const float4*)&attr[(long)s * DIM + j4];
  if (j4 >= NODEF) {
    float4 ev = *(const float4*)&edge_attr[e * EDGEF + (j4 - NODEF)];
    val.x += ev.x; val.y += ev.y; val.z += ev.z; val.w += ev.w;
  }
  float* out = &v[(long)d * DIM + j4];
  atomicAdd(out + 0, val.x);
  atomicAdd(out + 1, val.y);
  atomicAdd(out + 2, val.z);
  atomicAdd(out + 3, val.w);
}

// ---------------------------------------------------------------------------
// attr = v @ W_inner + b   ([N,256] x [256,256]) via bf16 WMMA.
// Block = 16 rows, 8 waves; each wave runs TWO independent accumulator
// chains (adjacent column tiles, shared A fragment) to overlap B-load
// latency with WMMA issue.
// ---------------------------------------------------------------------------
__global__ __launch_bounds__(256, 1)
void inner_gemm_kernel(const float* __restrict__ src, const __bf16* __restrict__ Wpk,
                       const float* __restrict__ bias, float* __restrict__ dst) {
  __shared__ __align__(32) __bf16 ldsA[16 * DIM];
  int tid = threadIdx.x;
  int wave = tid >> 5, lane = tid & 31;
  int row0 = blockIdx.x * 16;
  stage_A(src, row0, ldsA, tid);
  __syncthreads();

  v16bf Afrag[DIM / 32];
  #pragma unroll
  for (int kk = 0; kk < DIM / 32; ++kk)
    Afrag[kk] = *(const v16bf*)&ldsA[(kk * 32 + lane) << 4];

  int r_hi = (lane >> 4) << 3;
  int c_lo = lane & 15;
  int t0 = wave * 2;                                   // tiles t0, t0+1
  v8f acc0 = {}, acc1 = {};
  const __bf16* B0 = Wpk + (((long)t0 * (DIM / 32)) << 9) + (lane << 4);
  const __bf16* B1 = B0 + ((long)(DIM / 32) << 9);
  #pragma unroll
  for (int kk = 0; kk < DIM / 32; ++kk) {
    v16bf b0 = *(const v16bf*)&B0[(long)kk << 9];
    v16bf b1 = *(const v16bf*)&B1[(long)kk << 9];
    acc0 = WMMA_BF16(Afrag[kk], b0, acc0);
    acc1 = WMMA_BF16(Afrag[kk], b1, acc1);
  }
  int c0 = t0 * 16 + c_lo;
  float bv0 = bias[c0], bv1 = bias[c0 + 16];
  #pragma unroll
  for (int vr = 0; vr < 8; ++vr) {
    dst[(long)(row0 + r_hi + vr) * DIM + c0]      = acc0[vr] + bv0;
    dst[(long)(row0 + r_hi + vr) * DIM + c0 + 16] = acc1[vr] + bv1;
  }
}

// ---------------------------------------------------------------------------
// fp += sum_rows softmax(attr @ W_out + b).  Block = 16 rows; full 16x2048
// logits tile in LDS (128KB + 8KB A tile, fits 320KB WGP LDS).  Each wave
// owns 16 contiguous column tiles, processed as 8 dual-chain pairs, with a
// prefetch of the next pair's packed-B block.
// ---------------------------------------------------------------------------
__global__ __launch_bounds__(256, 1)
void fp_kernel(const float* __restrict__ attr, const __bf16* __restrict__ Wpk,
               const float* __restrict__ bias, float* __restrict__ fp) {
  __shared__ __align__(32) __bf16 ldsA[16 * DIM];
  __shared__ float logits[16 * LLD];
  int tid = threadIdx.x;
  int wave = tid >> 5, lane = tid & 31;
  int row0 = blockIdx.x * 16;
  stage_A(attr, row0, ldsA, tid);
  __syncthreads();

  v16bf Afrag[DIM / 32];
  #pragma unroll
  for (int kk = 0; kk < DIM / 32; ++kk)
    Afrag[kk] = *(const v16bf*)&ldsA[(kk * 32 + lane) << 4];

  int r_hi = (lane >> 4) << 3;
  int c_lo = lane & 15;
  const long tile_stride = (long)(DIM / 32) << 9;      // bf16 elems per tile
  for (int p = 0; p < 8; ++p) {                        // 8 pairs of tiles
    int t0 = wave * 16 + p * 2;
    const __bf16* B0 = Wpk + (long)t0 * tile_stride + (lane << 4);
    const __bf16* B1 = B0 + tile_stride;
    // prefetch next pair's packed-B block into cache (global_prefetch_b8)
    __builtin_prefetch(B0 + 2 * tile_stride, 0, 3);
    v8f acc0 = {}, acc1 = {};
    #pragma unroll
    for (int kk = 0; kk < DIM / 32; ++kk) {
      v16bf b0 = *(const v16bf*)&B0[(long)kk << 9];
      v16bf b1 = *(const v16bf*)&B1[(long)kk << 9];
      acc0 = WMMA_BF16(Afrag[kk], b0, acc0);
      acc1 = WMMA_BF16(Afrag[kk], b1, acc1);
    }
    int c0 = t0 * 16 + c_lo;
    float bv0 = bias[c0], bv1 = bias[c0 + 16];
    #pragma unroll
    for (int vr = 0; vr < 8; ++vr) {
      logits[(r_hi + vr) * LLD + c0]      = acc0[vr] + bv0;
      logits[(r_hi + vr) * LLD + c0 + 16] = acc1[vr] + bv1;
    }
  }
  __syncthreads();

  // Per-row softmax: wave w owns rows {w, w+8}; wave32 shfl reductions.
  #pragma unroll
  for (int rr = 0; rr < 2; ++rr) {
    int r = wave + rr * 8;
    float* row = &logits[r * LLD];
    float mx = -3.4e38f;
    for (int c = lane; c < OUTF; c += 32) mx = fmaxf(mx, row[c]);
    #pragma unroll
    for (int off = 16; off > 0; off >>= 1) mx = fmaxf(mx, __shfl_xor(mx, off, 32));
    float s = 0.f;
    for (int c = lane; c < OUTF; c += 32) {
      float e = __expf(row[c] - mx);
      row[c] = e;
      s += e;
    }
    #pragma unroll
    for (int off = 16; off > 0; off >>= 1) s += __shfl_xor(s, off, 32);
    float inv = 1.0f / s;
    for (int c = lane; c < OUTF; c += 32) row[c] *= inv;
  }
  __syncthreads();

  // Column sums over the 16 rows, one atomic per column per block.
  for (int c = tid; c < OUTF; c += 256) {
    float s = 0.f;
    #pragma unroll
    for (int r = 0; r < 16; ++r) s += logits[r * LLD + c];
    atomicAdd(&fp[c], s);
  }
}

// ---------------------------------------------------------------------------
extern "C" void kernel_launch(void* const* d_in, const int* in_sizes, int n_in,
                              void* d_out, int out_size, void* d_ws, size_t ws_size,
                              hipStream_t stream) {
  const float* node_attr = (const float*)d_in[0];
  const float* edge_attr = (const float*)d_in[1];
  const int*   esrc      = (const int*)d_in[2];
  const int*   edst      = (const int*)d_in[3];
  const float* W_inner   = (const float*)d_in[4];
  const float* b_inner   = (const float*)d_in[5];
  const float* W_output  = (const float*)d_in[6];
  const float* b_output  = (const float*)d_in[7];
  float* fp = (float*)d_out;

  const size_t ATTR_BYTES = (size_t)Nn * DIM * sizeof(float);        // 51.2 MB
  const size_t WI_BYTES   = (size_t)NDEPTH * DIM * DIM * sizeof(__bf16);
  char* ws = (char*)d_ws;
  float*  attr  = (float*)ws;
  float*  vbuf  = (float*)(ws + ATTR_BYTES);
  __bf16* Wi_pk = (__bf16*)(ws + 2 * ATTR_BYTES);
  __bf16* Wo_pk = (__bf16*)(ws + 2 * ATTR_BYTES + WI_BYTES);

  const long n4 = (long)Nn * (DIM / 4);                 // float4 count
  const int copy_blocks = (int)((n4 + 255) / 256);
  const int row_blocks  = Nn / 16;                      // 3125
  const int edge_blocks = (int)(((long)Mm * 64 + 255) / 256);

  // d_out accumulator starts at zero every call
  zero_kernel<<<(OUTF + 255) / 256, 256, 0, stream>>>(fp, OUTF);

  // attr = [node_attr | 0]
  pad_attr_kernel<<<copy_blocks, 256, 0, stream>>>(node_attr, attr);

  // pack weights to bf16 WMMA-B layout
  {
    long tot_i = (long)NDEPTH * (DIM / 16) * (DIM / 32) * 512;
    long tot_o = (long)(NDEPTH + 1) * (OUTF / 16) * (DIM / 32) * 512;
    pack_weight_bf16<<<(int)((tot_i + 255) / 256), 256, 0, stream>>>(
        W_inner, Wi_pk, DIM, DIM, NDEPTH);
    pack_weight_bf16<<<(int)((tot_o + 255) / 256), 256, 0, stream>>>(
        W_output, Wo_pk, DIM, OUTF, NDEPTH + 1);
  }

  // depth-0 fingerprint contribution
  fp_kernel<<<row_blocks, 256, 0, stream>>>(attr, Wo_pk, b_output, fp);

  for (int d = 0; d < NDEPTH; ++d) {
    // v = attr (then scatter-add messages into it)
    copy_f32x4_kernel<<<copy_blocks, 256, 0, stream>>>(attr, vbuf, n4);
    scatter_edges_kernel<<<edge_blocks, 256, 0, stream>>>(attr, edge_attr,
                                                          esrc, edst, vbuf);
    // attr = v @ W_inner[d] + b_inner[d]
    inner_gemm_kernel<<<row_blocks, 256, 0, stream>>>(
        vbuf, Wi_pk + (size_t)d * DIM * DIM, b_inner + (size_t)d * DIM, attr);
    // fp += softmax contribution for depth d+1
    fp_kernel<<<row_blocks, 256, 0, stream>>>(
        attr, Wo_pk + (size_t)(d + 1) * DIM * OUTF,
        b_output + (size_t)(d + 1) * OUTF, fp);
  }
}
